// DiffPoolLayer_26706106646695
// MI455X (gfx1250) — matrix operand order repaired and checked by reading, not compiled, not run on hardware
//
#include <hip/hip_runtime.h>
#include <stdint.h>

// DiffPool for MI455X (gfx1250), wave32 + WMMA bf16 (+ async global->LDS staging).
#define NN    64000
#define EE    1024000
#define DD    512
#define KTOTC 2048
#define BATCH 16
#define NPER  4000      // nodes per graph
#define KPER  128       // clusters per graph

typedef __attribute__((ext_vector_type(16))) __bf16 v16bf;
typedef __attribute__((ext_vector_type(8)))  float  v8f;

// Async global->LDS (GLOBAL_LOAD_ASYNC_TO_LDS_B128, tracked by ASYNCcnt).
// Probe result (round 2 diagnostic): arg0 is AS1 pointer to int __vector_size__(16).
#if defined(__gfx1250__) && __has_builtin(__builtin_amdgcn_global_load_async_to_lds_b128) && \
    __has_builtin(__builtin_amdgcn_s_wait_asynccnt)
#define HAVE_ASYNC_LDS 1
typedef int a4i __attribute__((__vector_size__(16)));
typedef __attribute__((address_space(1))) a4i ga4i_t;   // global (prints as "__device__")
typedef __attribute__((address_space(3))) a4i la4i_t;   // LDS   (prints as "__shared__")
#else
#define HAVE_ASYNC_LDS 0
#endif

// ---------------------------------------------------------------------------
// WMMA helpers (layouts per CDNA5 ISA §7.12.2)
// A 16x32 bf16: lanes 0-15 row M=lane, K={0..7,16..23}; lanes 16-31 row M=lane-16, K={8..15,24..31}
// B 32x16 bf16 staged transposed in LDS as ldsB[col*32 + k]: lanes 0-15 K={0..7,16..23}, lanes 16-31 K={8..15,24..31}
// ---------------------------------------------------------------------------
static __device__ __forceinline__ v16bf load_a_frag(const __bf16* ldsA, int row0, int lane) {
  const int l = lane & 15, hi = lane >> 4;
  const __bf16* p = ldsA + (row0 + l) * 32;
  v16bf a;
#pragma unroll
  for (int i = 0; i < 8; ++i) a[i] = p[hi * 8 + i];
#pragma unroll
  for (int i = 0; i < 8; ++i) a[8 + i] = p[16 + hi * 8 + i];
  return a;
}

static __device__ __forceinline__ v16bf load_b_frag(const __bf16* ldsB, int col0, int lane) {
  const int c = col0 + (lane & 15), hi = lane >> 4;
  const __bf16* p = ldsB + c * 32 + hi * 8;
  v16bf b;
#pragma unroll
  for (int i = 0; i < 8; ++i) { b[i] = p[i]; b[8 + i] = p[16 + i]; }
  return b;
}

static __device__ __forceinline__ v8f wmma_bf16(v16bf a, v16bf b, v8f c) {
  return __builtin_amdgcn_wmma_f32_16x16x32_bf16(false, a, false, b, (short)0, c, false, false);
}

static __device__ __forceinline__ v8f vzero8() {
  v8f z = {0.f, 0.f, 0.f, 0.f, 0.f, 0.f, 0.f, 0.f};
  return z;
}

// ---------------------------------------------------------------------------
// 1) Edge scatter: agg[dst] += feat[src]; deg[dst] += 1
// ---------------------------------------------------------------------------
__global__ __launch_bounds__(256) void k_scatter_agg(const int* __restrict__ src,
                                                     const int* __restrict__ dst,
                                                     const float* __restrict__ feat,
                                                     float* __restrict__ agg,
                                                     float* __restrict__ deg) {
  const int e = blockIdx.x * 2 + (threadIdx.x >> 7);
  const int q = threadIdx.x & 127;                     // 128 lanes * float4 = 512
  const int i = src[e], j = dst[e];
  const float4 v = ((const float4*)(feat + (size_t)i * DD))[q];
  float* a = agg + (size_t)j * DD + q * 4;
  atomicAdd(a + 0, v.x); atomicAdd(a + 1, v.y);
  atomicAdd(a + 2, v.z); atomicAdd(a + 3, v.w);
  if (q == 0) atomicAdd(deg + j, 1.0f);
}

// ---------------------------------------------------------------------------
// 2) xsum = bf16(feat + agg / max(deg,1))   packed 2 x bf16 per u32
// ---------------------------------------------------------------------------
__global__ __launch_bounds__(256) void k_finalize(const float* __restrict__ feat,
                                                  const float* __restrict__ agg,
                                                  const float* __restrict__ deg,
                                                  uint32_t* __restrict__ xsum2) {
  const int id = blockIdx.x * 256 + threadIdx.x;       // over NN*DD/2
  const int row = id >> 8;                             // DD/2 = 256 pairs per row
  const float inv = 1.0f / fmaxf(deg[row], 1.0f);
  const float2 f = ((const float2*)feat)[id];
  const float2 a = ((const float2*)agg)[id];
  const __bf16 h0 = (__bf16)(f.x + a.x * inv);
  const __bf16 h1 = (__bf16)(f.y + a.y * inv);
  const uint32_t u0 = __builtin_bit_cast(unsigned short, h0);
  const uint32_t u1 = __builtin_bit_cast(unsigned short, h1);
  xsum2[id] = u0 | (u1 << 16);
}

// ---------------------------------------------------------------------------
// A-tile stage helper: copy 128 rows x 32 bf16 (64B/row) from row-major bf16
// global (row stride DD*2 bytes) into ldsA, async when available.
// rows with r >= nvalid are skipped (LDS pre-zeroed by caller when ragged).
// ---------------------------------------------------------------------------
static __device__ __forceinline__ void stage_a_tile(const uint32_t* __restrict__ gsrc_u32,
                                                    uint32_t* __restrict__ ldsA32,
                                                    int tid, int nvalid) {
#if HAVE_ASYNC_LDS
#pragma unroll
  for (int i = 0; i < 2; ++i) {                        // 512 chunks of 16B, 2 per thread
    const int ch = tid + i * 256;
    const int r = ch >> 2, q = ch & 3;                 // 4 chunks per 64B row
    if (r < nvalid) {
      __builtin_amdgcn_global_load_async_to_lds_b128(
          (ga4i_t*)((const char*)gsrc_u32 + (size_t)r * (DD * 2) + q * 16),
          (la4i_t*)((char*)ldsA32 + r * 64 + q * 16), 0, 0);
    }
  }
  __builtin_amdgcn_s_wait_asynccnt(0);
#else
#pragma unroll
  for (int i = 0; i < 8; ++i) {                        // 2048 u32, 8 per thread
    const int idx = tid + i * 256;
    const int r = idx >> 4, kk = idx & 15;             // 16 u32 per row
    if (r < nvalid) ldsA32[r * 16 + kk] = gsrc_u32[(size_t)r * 256 + kk];
  }
#endif
}

// ---------------------------------------------------------------------------
// 3) hfeat = bf16(relu(xsum @ W_feat + b_feat))   [NN x DD], tiles 128x128, K step 32
// ---------------------------------------------------------------------------
__global__ __launch_bounds__(256) void k_gemm_feat(const uint32_t* __restrict__ xsum2,
                                                   const float* __restrict__ W,
                                                   const float* __restrict__ bias,
                                                   __bf16* __restrict__ out) {
  __shared__ __bf16 ldsA[128 * 32];
  __shared__ __bf16 ldsB[128 * 32];
  uint32_t* ldsA32 = (uint32_t*)ldsA;

  const int tid = threadIdx.x;
  const int row0 = (blockIdx.x >> 2) * 128;
  const int col0 = (blockIdx.x & 3) * 128;
  const int wave = tid >> 5, lane = tid & 31;
  const int wm = wave & 3, wn = wave >> 2;

  v8f acc[2][4];
#pragma unroll
  for (int a = 0; a < 2; ++a)
#pragma unroll
    for (int b = 0; b < 4; ++b) acc[a][b] = vzero8();

  for (int kb = 0; kb < DD; kb += 32) {
    stage_a_tile(xsum2 + (size_t)row0 * 256 + (kb >> 1), ldsA32, tid, 128);
#pragma unroll
    for (int i = 0; i < 16; ++i) {                     // B: 4096 bf16 (transposed)
      const int idx = tid + i * 256;
      const int k = idx >> 7, c = idx & 127;
      ldsB[c * 32 + k] = (__bf16)W[(size_t)(kb + k) * DD + col0 + c];
    }
    __builtin_prefetch(&W[(size_t)(kb + 32) * DD + col0], 0, 1);
    __syncthreads();

    v16bf afr[2], bfr[4];
#pragma unroll
    for (int tm = 0; tm < 2; ++tm) afr[tm] = load_a_frag(ldsA, wm * 32 + tm * 16, lane);
#pragma unroll
    for (int tn = 0; tn < 4; ++tn) bfr[tn] = load_b_frag(ldsB, wn * 64 + tn * 16, lane);
#pragma unroll
    for (int tm = 0; tm < 2; ++tm)
#pragma unroll
      for (int tn = 0; tn < 4; ++tn) acc[tm][tn] = wmma_bf16(afr[tm], bfr[tn], acc[tm][tn]);
    __syncthreads();
  }

  const int lane16 = lane & 15, laneh = lane >> 4;
#pragma unroll
  for (int tm = 0; tm < 2; ++tm)
#pragma unroll
    for (int tn = 0; tn < 4; ++tn) {
      const int col = col0 + wn * 64 + tn * 16 + lane16;
      const float bb = bias[col];
      const int rb = row0 + wm * 32 + tm * 16 + laneh * 8;
#pragma unroll
      for (int j = 0; j < 8; ++j) {
        float v = acc[tm][tn][j] + bb;
        out[(size_t)(rb + j) * DD + col] = (__bf16)(v > 0.f ? v : 0.f);
      }
    }
}

// ---------------------------------------------------------------------------
// 4) per-graph assign logits: s = xsum_g @ W_pool[:, g*128 : g*128+128] + b_pool  (f32, pre-softmax)
//    grid = BATCH * 32 row-tiles (4000 = 31*128 + 32, ragged tail: LDS rows pre-zeroed once)
// ---------------------------------------------------------------------------
__global__ __launch_bounds__(256) void k_gemm_assign(const uint32_t* __restrict__ xsum2,
                                                     const float* __restrict__ Wp,
                                                     const float* __restrict__ bp,
                                                     float* __restrict__ sdst) {
  __shared__ __bf16 ldsA[128 * 32];
  __shared__ __bf16 ldsB[128 * 32];
  uint32_t* ldsA32 = (uint32_t*)ldsA;

  const int tid = threadIdx.x;
  const int g = blockIdx.x >> 5;
  const int row0 = (blockIdx.x & 31) * 128;            // row within graph
  const int nvalid = NPER - row0 < 128 ? NPER - row0 : 128;  // 128 or 32
  const int wave = tid >> 5, lane = tid & 31;
  const int wm = wave & 3, wn = wave >> 2;

  // pre-zero ragged A rows once; K-loop staging then only touches valid rows
  if (nvalid < 128) {
#pragma unroll
    for (int i = 0; i < 8; ++i) {
      const int idx = tid + i * 256;
      if ((idx >> 4) >= nvalid) ldsA32[idx] = 0u;
    }
  }
  __syncthreads();

  v8f acc[2][4];
#pragma unroll
  for (int a = 0; a < 2; ++a)
#pragma unroll
    for (int b = 0; b < 4; ++b) acc[a][b] = vzero8();

  for (int kb = 0; kb < DD; kb += 32) {
    stage_a_tile(xsum2 + (size_t)(g * NPER + row0) * 256 + (kb >> 1), ldsA32, tid, nvalid);
#pragma unroll
    for (int i = 0; i < 16; ++i) {
      const int idx = tid + i * 256;
      const int k = idx >> 7, c = idx & 127;
      ldsB[c * 32 + k] = (__bf16)Wp[(size_t)(kb + k) * KTOTC + g * KPER + c];
    }
    __builtin_prefetch(&Wp[(size_t)(kb + 32) * KTOTC + g * KPER], 0, 1);
    __syncthreads();

    v16bf afr[2], bfr[4];
#pragma unroll
    for (int tm = 0; tm < 2; ++tm) afr[tm] = load_a_frag(ldsA, wm * 32 + tm * 16, lane);
#pragma unroll
    for (int tn = 0; tn < 4; ++tn) bfr[tn] = load_b_frag(ldsB, wn * 64 + tn * 16, lane);
#pragma unroll
    for (int tm = 0; tm < 2; ++tm)
#pragma unroll
      for (int tn = 0; tn < 4; ++tn) acc[tm][tn] = wmma_bf16(afr[tm], bfr[tn], acc[tm][tn]);
    __syncthreads();
  }

  const int lane16 = lane & 15, laneh = lane >> 4;
#pragma unroll
  for (int tm = 0; tm < 2; ++tm)
#pragma unroll
    for (int tn = 0; tn < 4; ++tn) {
      const int cc = wn * 64 + tn * 16 + lane16;       // col within 128-block
      const float bb = bp[g * KPER + cc];
      const int rb = row0 + wm * 32 + tm * 16 + laneh * 8;
#pragma unroll
      for (int j = 0; j < 8; ++j) {
        const int r = rb + j;
        if (r < NPER)
          sdst[(size_t)(g * NPER + r) * KPER + cc] = acc[tm][tn][j] + bb;
      }
    }
}

// ---------------------------------------------------------------------------
// 5) masked softmax per node over its 128-block; reproduces reference's
//    softmax(assign*mask)*mask renorm: denom = sum_in + 1e-13*(sum_in + 1920*exp(-m))
// ---------------------------------------------------------------------------
__global__ __launch_bounds__(256) void k_softmax(float* __restrict__ s) {
  const int row = blockIdx.x * 8 + (threadIdx.x >> 5);
  const int lane = threadIdx.x & 31;
  float4 v = ((const float4*)(s + (size_t)row * KPER))[lane];
  float m = fmaxf(fmaxf(v.x, v.y), fmaxf(v.z, v.w));
  m = fmaxf(m, 0.0f);                                  // off-block zeros join the max
#pragma unroll
  for (int off = 16; off > 0; off >>= 1) m = fmaxf(m, __shfl_xor(m, off, 32));
  const float e0 = __expf(v.x - m), e1 = __expf(v.y - m);
  const float e2 = __expf(v.z - m), e3 = __expf(v.w - m);
  float sum = e0 + e1 + e2 + e3;
#pragma unroll
  for (int off = 16; off > 0; off >>= 1) sum += __shfl_xor(sum, off, 32);
  const float zall = sum + 1920.0f * __expf(-m);       // (KTOT-KPER) masked exp(0-m) terms
  const float inv = 1.0f / (sum + 1e-13f * zall);
  float4 o = {e0 * inv, e1 * inv, e2 * inv, e3 * inv};
  ((float4*)(s + (size_t)row * KPER))[lane] = o;
}

// ---------------------------------------------------------------------------
// 6) adj_s[src] (block of graph(dst)) += s[dst]   (f32 atomics, [NN x KTOT])
// ---------------------------------------------------------------------------
__global__ __launch_bounds__(256) void k_scatter_adjs(const int* __restrict__ src,
                                                      const int* __restrict__ dst,
                                                      const float* __restrict__ s,
                                                      float* __restrict__ adjs) {
  const int e = blockIdx.x * 2 + (threadIdx.x >> 7);
  const int c = threadIdx.x & 127;
  const int i = src[e], j = dst[e];
  const int gj = j / NPER;
  const float v = s[(size_t)j * KPER + c];
  atomicAdd(adjs + (size_t)i * KTOTC + gj * KPER + c, v);
}

// ---------------------------------------------------------------------------
// 7) per-graph sT GEMM: out[g*128 + m][col] = sum_r s[g*4000+r][m] * B[g*4000+r][col]
//    BF32: B is f32 (adj_s, ncols=2048, out=adj_new) else bf16 (hfeat, ncols=512, out=h)
//    K = 4000 = 125 * 32 exact; A staged transposed f32->bf16
// ---------------------------------------------------------------------------
template <bool BF32>
__global__ __launch_bounds__(256) void k_gemm_sT(const float* __restrict__ s,
                                                 const void* __restrict__ Bp,
                                                 float* __restrict__ out,
                                                 int ncols) {
  __shared__ __bf16 ldsA[128 * 32];
  __shared__ __bf16 ldsB[128 * 32];

  const int tid = threadIdx.x;
  const int tilesN = ncols >> 7;
  const int g = blockIdx.x / tilesN;
  const int col0 = (blockIdx.x % tilesN) * 128;
  const int nodeBase = g * NPER;
  const int wave = tid >> 5, lane = tid & 31;
  const int wm = wave & 3, wn = wave >> 2;

  v8f acc[2][4];
#pragma unroll
  for (int a = 0; a < 2; ++a)
#pragma unroll
    for (int b = 0; b < 4; ++b) acc[a][b] = vzero8();

  for (int kb = 0; kb < NPER; kb += 32) {
#pragma unroll
    for (int i = 0; i < 16; ++i) {                     // A: s transposed, 4096 elems
      const int idx = tid + i * 256;
      const int k = idx >> 7, mm = idx & 127;
      ldsA[mm * 32 + k] = (__bf16)s[(size_t)(nodeBase + kb + k) * KPER + mm];
    }
#pragma unroll
    for (int i = 0; i < 16; ++i) {                     // B: transposed stage
      const int idx = tid + i * 256;
      const int k = idx >> 7, c = idx & 127;
      if constexpr (BF32) {
        ldsB[c * 32 + k] =
            (__bf16)((const float*)Bp)[(size_t)(nodeBase + kb + k) * ncols + col0 + c];
      } else {
        ldsB[c * 32 + k] =
            ((const __bf16*)Bp)[(size_t)(nodeBase + kb + k) * ncols + col0 + c];
      }
    }
    __builtin_prefetch(&s[(size_t)(nodeBase + kb + 32) * KPER], 0, 1);
    __syncthreads();

    v16bf afr[2], bfr[4];
#pragma unroll
    for (int tm = 0; tm < 2; ++tm) afr[tm] = load_a_frag(ldsA, wm * 32 + tm * 16, lane);
#pragma unroll
    for (int tn = 0; tn < 4; ++tn) bfr[tn] = load_b_frag(ldsB, wn * 64 + tn * 16, lane);
#pragma unroll
    for (int tm = 0; tm < 2; ++tm)
#pragma unroll
      for (int tn = 0; tn < 4; ++tn) acc[tm][tn] = wmma_bf16(afr[tm], bfr[tn], acc[tm][tn]);
    __syncthreads();
  }

  const int lane16 = lane & 15, laneh = lane >> 4;
#pragma unroll
  for (int tm = 0; tm < 2; ++tm)
#pragma unroll
    for (int tn = 0; tn < 4; ++tn) {
      const int col = col0 + wn * 64 + tn * 16 + lane16;
      const int rb = wm * 32 + tm * 16 + laneh * 8;    // cluster row 0..127
#pragma unroll
      for (int j = 0; j < 8; ++j)
        out[(size_t)(g * KPER + rb + j) * ncols + col] = acc[tm][tn][j];
    }
}

// ---------------------------------------------------------------------------
// launch
// ---------------------------------------------------------------------------
extern "C" void kernel_launch(void* const* d_in, const int* in_sizes, int n_in,
                              void* d_out, int out_size, void* d_ws, size_t ws_size,
                              hipStream_t stream) {
  const float* feat   = (const float*)d_in[0];
  const float* W_feat = (const float*)d_in[1];
  const float* b_feat = (const float*)d_in[2];
  const float* W_pool = (const float*)d_in[3];
  const float* b_pool = (const float*)d_in[4];
  const int*   src    = (const int*)d_in[5];
  const int*   dst    = (const int*)d_in[6];

  // Workspace layout (~656 MB total; adj_s region time-shares with agg/deg):
  //   [0, 524288000)          : agg[NN*DD] + deg[NN]  (early)  /  adj_s[NN*KTOT] (late)
  //   then xsum bf16 [NN*DD], hfeat bf16 [NN*DD], s f32 [NN*KPER]
  char* ws = (char*)d_ws;
  const size_t SCR = (size_t)NN * KTOTC * sizeof(float);
  float*  agg   = (float*)ws;
  float*  deg   = (float*)(ws + (size_t)NN * DD * sizeof(float));
  float*  adjs  = (float*)ws;
  __bf16* xsum  = (__bf16*)(ws + SCR);
  __bf16* hfeat = xsum + (size_t)NN * DD;
  float*  sbuf  = (float*)(hfeat + (size_t)NN * DD);

  float* out_adj = (float*)d_out;                       // [KTOT][KTOT]
  float* out_h   = out_adj + (size_t)KTOTC * KTOTC;     // [KTOT][DD]

  (void)hipMemsetAsync(agg, 0, ((size_t)NN * DD + NN) * sizeof(float), stream);
  k_scatter_agg<<<EE / 2, 256, 0, stream>>>(src, dst, feat, agg, deg);
  k_finalize<<<(NN * DD / 2) / 256, 256, 0, stream>>>(feat, agg, deg, (uint32_t*)xsum);

  k_gemm_feat<<<(NN / 128) * (DD / 128), 256, 0, stream>>>((const uint32_t*)xsum,
                                                           W_feat, b_feat, hfeat);
  k_gemm_assign<<<BATCH * 32, 256, 0, stream>>>((const uint32_t*)xsum, W_pool, b_pool, sbuf);
  k_softmax<<<NN / 8, 256, 0, stream>>>(sbuf);

  (void)hipMemsetAsync(adjs, 0, (size_t)NN * KTOTC * sizeof(float), stream);
  k_scatter_adjs<<<EE / 2, 256, 0, stream>>>(src, dst, sbuf, adjs);

  k_gemm_sT<false><<<BATCH * (DD / 128), 256, 0, stream>>>(sbuf, hfeat, out_h, DD);
  k_gemm_sT<true><<<BATCH * (KTOTC / 128), 256, 0, stream>>>(sbuf, adjs, out_adj, KTOTC);

  (void)in_sizes; (void)n_in; (void)out_size; (void)ws_size;
}
